// SelectiveSSM_20461224198830
// MI455X (gfx1250) — compile-verified, matrix operand order checked by reading, not run on hardware
//
#include <hip/hip_runtime.h>
#include <hip/hip_bf16.h>
#include <math.h>

// ---------------------------------------------------------------------------
// Selective SSM (Mamba-style), fully fused for MI455X (gfx1250, wave32).
//   Bsz=2, L=2048, D=768, N=16.
//   Phase A (tiny): swizzle fp32 weights [W_delta; W_B; W_C] and x into bf16
//                   WMMA operand layout in d_ws  (~45.2 MB workspace).
//   Phase B (hot):  48x2 workgroups, 16 waves each. Wave w owns channel d0+w:
//                   WMMA-projects B/C (wave0 additionally: delta) for 64
//                   timesteps, then scans them in-register via lane shuffles.
//   Wave-uniform branches are forced scalar (readfirstlane) so every WMMA
//   issues with EXEC == all-ones as the ISA requires.
// ---------------------------------------------------------------------------

typedef __attribute__((ext_vector_type(16))) __bf16 v16bf;
typedef __attribute__((ext_vector_type(8)))  float  v8f;

#define BSZ 2
#define LSEQ 2048
#define DM 768
#define NST 16
#define KS 24                      // 768 / 32 K-slices per WMMA chain
#define MT 4                       // row-tiles (16 timesteps each) per chunk
#define LT (LSEQ / 16)             // 128 row tiles over L
#define NCHUNK (LSEQ / (16 * MT))  // 32 chunks
#define NTILES 1584                // 48 delta + 768 B + 768 C column tiles
#define WSW_ELEMS ((size_t)NTILES * KS * 512)       // 19,464,192 bf16
#define XSW_ELEMS ((size_t)BSZ * LT * KS * 512)     //  3,145,728 bf16
// total workspace: (WSW_ELEMS + XSW_ELEMS) * 2 B = 45,219,840 bytes

static __device__ __forceinline__ unsigned short f2bf(float f) {
  union { float f; unsigned u; } v; v.f = f;
  unsigned r = v.u + 0x7FFFu + ((v.u >> 16) & 1u);   // round-to-nearest-even
  return (unsigned short)(r >> 16);
}

// WMMA 16x16x32 bf16 -> f32 accumulate (8-arg form, probe-confirmed).
static __device__ __forceinline__ v8f wmma_bf(v16bf a, v16bf b, v8f c) {
  return __builtin_amdgcn_wmma_f32_16x16x32_bf16(false, a, false, b,
                                                 (short)0, c, false, false);
}

// Uniform broadcast of lane `l`'s value -> v_readlane (VALU, no LDS pipe).
static __device__ __forceinline__ float bcast_lane(float v, int l) {
  return __builtin_bit_cast(float,
      __builtin_amdgcn_readlane(__builtin_bit_cast(int, v), l));
}

// ---------------------------------------------------------------------------
// Swizzle weights -> bf16 B-operand layout.
// Per (tile, kslice) blob: 32 lanes x 16 bf16; lane(0..15)=n with
// K {0..7,16..23}, lane(16..31)=n with K {8..15,24..31}  (ISA 7.12.2).
// Tile map: [0,48) -> W_delta, [48,816) -> W_B, [816,1584) -> W_C.
// ---------------------------------------------------------------------------
__global__ __launch_bounds__(32) void swz_weights(
    const float* __restrict__ Wd, const float* __restrict__ Wb,
    const float* __restrict__ Wc, unsigned short* __restrict__ dst) {
  const int tile = blockIdx.x / KS;
  const int s    = blockIdx.x % KS;
  const int lane = threadIdx.x;
  const float* src; int row0;
  if (tile < 48)       { src = Wd; row0 = tile * 16; }
  else if (tile < 816) { src = Wb; row0 = (tile - 48) * 16; }
  else                 { src = Wc; row0 = (tile - 816) * 16; }
  const int n     = lane & 15;
  const int kbase = (lane >> 4) * 8;
  const float* rowp = src + (size_t)(row0 + n) * DM + s * 32;
  unsigned pk[8];
#pragma unroll
  for (int j = 0; j < 8; ++j) {
    const int i0 = 2 * j, i1 = 2 * j + 1;
    const int k0 = (i0 < 8) ? (kbase + i0) : (16 + kbase + (i0 - 8));
    const int k1 = (i1 < 8) ? (kbase + i1) : (16 + kbase + (i1 - 8));
    pk[j] = (unsigned)f2bf(rowp[k0]) | ((unsigned)f2bf(rowp[k1]) << 16);
  }
  uint4* out4 = (uint4*)(dst + ((size_t)blockIdx.x * 32 + lane) * 16);
  out4[0] = make_uint4(pk[0], pk[1], pk[2], pk[3]);
  out4[1] = make_uint4(pk[4], pk[5], pk[6], pk[7]);
}

// ---------------------------------------------------------------------------
// Swizzle x -> bf16 A-operand layout.  blockIdx.x = (b*LT + lt)*KS + s.
// ---------------------------------------------------------------------------
__global__ __launch_bounds__(32) void swz_x(
    const float* __restrict__ x, unsigned short* __restrict__ dst) {
  const int s    = blockIdx.x % KS;
  const int blt  = blockIdx.x / KS;
  const int lt   = blt % LT;
  const int b    = blt / LT;
  const int lane = threadIdx.x;
  const int m     = lane & 15;
  const int kbase = (lane >> 4) * 8;
  const float* rowp = x + ((size_t)b * LSEQ + lt * 16 + m) * DM + s * 32;
  unsigned pk[8];
#pragma unroll
  for (int j = 0; j < 8; ++j) {
    const int i0 = 2 * j, i1 = 2 * j + 1;
    const int k0 = (i0 < 8) ? (kbase + i0) : (16 + kbase + (i0 - 8));
    const int k1 = (i1 < 8) ? (kbase + i1) : (16 + kbase + (i1 - 8));
    pk[j] = (unsigned)f2bf(rowp[k0]) | ((unsigned)f2bf(rowp[k1]) << 16);
  }
  uint4* out4 = (uint4*)(dst + ((size_t)blockIdx.x * 32 + lane) * 16);
  out4[0] = make_uint4(pk[0], pk[1], pk[2], pk[3]);
  out4[1] = make_uint4(pk[4], pk[5], pk[6], pk[7]);
}

// ---------------------------------------------------------------------------
// Fused projection + scan.
// C/D tile layout (ISA 7.12.2): lane(0..15)=col n holds rows 0..7 in
// elements 0..7; lane(16..31)=col n holds rows 8..15. B/C tile columns are
// exactly (channel d, state n), so the scan runs directly on the tiles:
//   row m <  8 : lanes 0..15 already hold the value (no shuffle),
//   row m >= 8 : one half-swap (__shfl_xor 16).
// ---------------------------------------------------------------------------
__global__ __launch_bounds__(512) void ssm_fused(
    const float* __restrict__ x, const float* __restrict__ A,
    const float* __restrict__ dbias,
    const unsigned short* __restrict__ Wsw,
    const unsigned short* __restrict__ Xsw,
    float* __restrict__ out) {
  const int lane = threadIdx.x & 31;
  // Scalar (SGPR) wave index: forces s_cmp/s_cbranch, never EXEC masking.
  const int wave = __builtin_amdgcn_readfirstlane((int)threadIdx.x) >> 5;
  const int dblk = blockIdx.x;       // 0..47
  const int b    = blockIdx.y;       // 0..1
  const int d0   = dblk * 16;
  const int d    = d0 + wave;        // this wave's channel
  const int tB = 48 + dblk * 16 + wave;    // W_B column tile (rows d*16..+16)
  const int tC = 816 + dblk * 16 + wave;   // W_C column tile
  const int tD = dblk;                     // delta tile (wave 0 only)
  const v16bf* __restrict__ Wv = (const v16bf*)Wsw;
  const v16bf* __restrict__ Xv = (const v16bf*)Xsw;

  __shared__ float sDelta[MT * 16 * 16];   // [m_in_chunk][channel_in_block]

  const float a_dn = A[d * NST + (lane & 15)];     // A[d][n], n = lane%16
  const float bias = dbias[d0 + (lane & 15)];      // used by wave 0
  float h = 0.0f;                                  // scan state h[d][n]

  for (int c = 0; c < NCHUNK; ++c) {
    const int lt0 = c * MT;
    v8f accB[MT] = {}, accC[MT] = {};

    if (wave == 0) {
      // -------- wave 0: B, C and delta tiles (EXEC stays all-ones) --------
      v8f accD[MT] = {};
      for (int s = 0; s < KS; ++s) {
        v16bf wb = Wv[(size_t)(tB * KS + s) * 32 + lane];
        v16bf wc = Wv[(size_t)(tC * KS + s) * 32 + lane];
        v16bf wd = Wv[(size_t)(tD * KS + s) * 32 + lane];
        if (s + 1 < KS) {
          __builtin_prefetch(&Wv[(size_t)(tB * KS + s + 1) * 32 + lane], 0, 3);
          __builtin_prefetch(&Wv[(size_t)(tC * KS + s + 1) * 32 + lane], 0, 3);
        }
#pragma unroll
        for (int mt = 0; mt < MT; ++mt) {
          v16bf av = Xv[(size_t)(((b * LT) + lt0 + mt) * KS + s) * 32 + lane];
          accB[mt] = wmma_bf(av, wb, accB[mt]);
          accC[mt] = wmma_bf(av, wc, accC[mt]);
          accD[mt] = wmma_bf(av, wd, accD[mt]);
        }
      }
      // softplus(delta_logits + bias) -> LDS for all 16 channels.
      const int dd = lane & 15;          // channel in block (tile column)
      const int mh = (lane >> 4) * 8;    // row half offset
#pragma unroll
      for (int mt = 0; mt < MT; ++mt)
#pragma unroll
        for (int r = 0; r < 8; ++r) {
          float z  = accD[mt][r] + bias;
          float sp = (z > 20.0f) ? z : log1pf(__expf(z));
          sDelta[(mt * 16 + mh + r) * 16 + dd] = sp;
        }
    } else {
      // -------- waves 1..15: B and C tiles only --------
      for (int s = 0; s < KS; ++s) {
        v16bf wb = Wv[(size_t)(tB * KS + s) * 32 + lane];
        v16bf wc = Wv[(size_t)(tC * KS + s) * 32 + lane];
        if (s + 1 < KS) {
          __builtin_prefetch(&Wv[(size_t)(tB * KS + s + 1) * 32 + lane], 0, 3);
          __builtin_prefetch(&Wv[(size_t)(tC * KS + s + 1) * 32 + lane], 0, 3);
        }
#pragma unroll
        for (int mt = 0; mt < MT; ++mt) {
          v16bf av = Xv[(size_t)(((b * LT) + lt0 + mt) * KS + s) * 32 + lane];
          accB[mt] = wmma_bf(av, wb, accB[mt]);
          accC[mt] = wmma_bf(av, wc, accC[mt]);
        }
      }
    }
    __syncthreads();

    // Preload x[b, l, d] and delta[l, d] for the 64 chunk timesteps:
    // lane j (mod 16) holds timestep mt*16+j; broadcast per-step via readlane.
    float xv[MT], dl[MT];
#pragma unroll
    for (int mt = 0; mt < MT; ++mt) {
      const int m16 = lane & 15;
      const int l   = (lt0 + mt) * 16 + m16;
      xv[mt] = x[((size_t)b * LSEQ + l) * DM + d];
      dl[mt] = sDelta[(mt * 16 + m16) * 16 + wave];
    }

    // In-register scan. lanes 0..15 = state index n (upper half duplicates
    // the arithmetic but only lane 0 of the lower half stores).
#pragma unroll
    for (int mt = 0; mt < MT; ++mt) {
#pragma unroll
      for (int r = 0; r < 16; ++r) {
        float bsel, csel;
        if (r < 8) {              // rows 0..7 live in lanes 0..15 already
          bsel = accB[mt][r];
          csel = accC[mt][r];
        } else {                  // rows 8..15 live in lanes 16..31: half-swap
          bsel = __shfl_xor(accB[mt][r - 8], 16, 32);
          csel = __shfl_xor(accC[mt][r - 8], 16, 32);
        }
        float xm = bcast_lane(xv[mt], r);   // v_readlane broadcast
        float dm = bcast_lane(dl[mt], r);
        float dA   = dm * a_dn;             // < 0 always (A in (-1,-0.5))
        float abar = __expf(dA);
        float bx   = (abar - 1.0f) / dA * dm * bsel * xm;   // ZOH B_bar * x
        h = abar * h + bx;
        float p = h * csel;                 // y = sum_n h*C
        p += __shfl_xor(p, 8, 16);
        p += __shfl_xor(p, 4, 16);
        p += __shfl_xor(p, 2, 16);
        p += __shfl_xor(p, 1, 16);
        if (lane == 0) {
          const int l = (lt0 + mt) * 16 + r;
          out[((size_t)b * LSEQ + l) * DM + d] = p;
        }
      }
    }
    __syncthreads();   // sDelta reused next chunk
  }
}

// ---------------------------------------------------------------------------
extern "C" void kernel_launch(void* const* d_in, const int* in_sizes, int n_in,
                              void* d_out, int out_size, void* d_ws,
                              size_t ws_size, hipStream_t stream) {
  const float* x  = (const float*)d_in[0];   // (2,2048,768)
  const float* A  = (const float*)d_in[1];   // (768,16)
  const float* Wd = (const float*)d_in[2];   // (768,768)
  const float* db = (const float*)d_in[3];   // (768,)
  const float* Wb = (const float*)d_in[4];   // (12288,768)
  const float* Wc = (const float*)d_in[5];   // (12288,768)
  float* out = (float*)d_out;

  unsigned short* Wsw = (unsigned short*)d_ws;
  unsigned short* Xsw = Wsw + WSW_ELEMS;     // 32B-aligned (offset % 32 == 0)

  swz_weights<<<NTILES * KS, 32, 0, stream>>>(Wd, Wb, Wc, Wsw);
  swz_x<<<BSZ * LT * KS, 32, 0, stream>>>(x, Xsw);
  ssm_fused<<<dim3(DM / 16, BSZ), 512, 0, stream>>>(x, A, db, Wsw, Xsw, out);
}
